// MultiHeadAttention_18056042513155
// MI455X (gfx1250) — matrix operand order repaired and checked by reading, not compiled, not run on hardware
//
#include <hip/hip_runtime.h>

// ---------------------------------------------------------------------------
// MHA forward for MI455X (gfx1250, wave32) using V_WMMA_F32_16X16X32_BF16.
// B=2, S=2048, D=1024, H=16, Dh=64. Causal + pad mask, softmax scale 1/sqrt(S).
// ---------------------------------------------------------------------------

typedef __bf16 bf16;
typedef bf16 v16bf __attribute__((ext_vector_type(16)));
typedef bf16 v8bf  __attribute__((ext_vector_type(8)));
typedef float v8f  __attribute__((ext_vector_type(8)));

constexpr int   Bc  = 2;
constexpr int   Sc  = 2048;
constexpr int   Dc  = 1024;
constexpr int   Hc  = 16;
constexpr int   Dh  = 64;
constexpr float NEGV   = -1.0e9f;
constexpr float QSCALE = 0.022097086912079612f; // 1/sqrt(2048)

__device__ __forceinline__ v8f wmma_bf16(v16bf a, v16bf b, v8f c) {
  // D = A(16x32 bf16) * B(32x16 bf16) + C(16x16 f32)
  return __builtin_amdgcn_wmma_f32_16x16x32_bf16(false, a, false, b,
                                                 (short)0, c, false, false);
}

// ds_swizzle group-of-32 XOR patterns (and_mask=0x1f, xor in bits [14:10])
template<int PAT>
__device__ __forceinline__ float swz(float x) {
  int xi = __builtin_bit_cast(int, x);
  int r  = __builtin_amdgcn_ds_swizzle(xi, PAT);
  return __builtin_bit_cast(float, r);
}
__device__ __forceinline__ float redmax16(float x) {   // max across 16-lane half
  x = fmaxf(x, swz<0x041f>(x));  // SWAPX1
  x = fmaxf(x, swz<0x081f>(x));  // SWAPX2
  x = fmaxf(x, swz<0x101f>(x));  // SWAPX4
  x = fmaxf(x, swz<0x201f>(x));  // SWAPX8
  return x;
}
__device__ __forceinline__ float redsum16(float x) {   // sum across 16-lane half
  x += swz<0x041f>(x);
  x += swz<0x081f>(x);
  x += swz<0x101f>(x);
  x += swz<0x201f>(x);
  return x;
}

// A-fragment (16x32, bf16): lane = M (lane&15); K(i) = i + 16*half.
// `row` already points at (global row, k-chunk base). Two 16B runs per lane.
__device__ __forceinline__ v16bf load_a_bf16(const bf16* row, int half) {
  v8bf lo = *(const v8bf*)(row + 16 * half);
  v8bf hi = *(const v8bf*)(row + 8 + 16 * half);
  v16bf a;
  #pragma unroll
  for (int i = 0; i < 8; ++i) { a[i] = lo[i]; a[i + 8] = hi[i]; }
  return a;
}

// ---------------------------------------------------------------------------
// Kernel 0: transpose + convert weights to bf16.  in[b][r][c] -> out[b][c][r]
// ---------------------------------------------------------------------------
__global__ void mha_wprep(const float* __restrict__ in, bf16* __restrict__ out,
                          int rows, int cols, int batch) {
  int idx = blockIdx.x * blockDim.x + threadIdx.x;
  int total = batch * rows * cols;
  if (idx >= total) return;
  int per = rows * cols;
  int b = idx / per, rem = idx % per;
  int r = rem / cols, c = rem % cols;
  out[((size_t)b * cols + c) * rows + r] = (bf16)in[idx];
}

// ---------------------------------------------------------------------------
// Kernel 1: per-head projection GEMM. X[B][S][D] f32 @ WT[H][Dh][D] bf16.
// One wave: 16 rows x 64 cols, K loop over D in chunks of 32, software
// pipelined (prefetch chunk kk+1 while WMMAs of chunk kk execute).
// TR=0 -> out[b][h][s][dh];  TR=1 -> out[b][h][dh][s]  (for V).
// ---------------------------------------------------------------------------
template<int TR>
__global__ __launch_bounds__(32)
void mha_proj(const float* __restrict__ X, const bf16* __restrict__ WT,
              bf16* __restrict__ out, float scale) {
  const int wid  = blockIdx.x;              // b * H * (S/16) waves
  const int MT   = Sc / 16;                 // 128
  const int mt   = wid % MT;
  const int h    = (wid / MT) % Hc;
  const int b    = wid / (MT * Hc);
  const int lane = threadIdx.x;
  const int n16  = lane & 15;
  const int half = lane >> 4;

  const float* arow  = X + ((size_t)b * Sc + mt * 16 + n16) * Dc;
  const bf16*  wbase = WT + ((size_t)h * Dh + n16) * Dc + 16 * half;

  auto loadA = [&](int kk) -> v16bf {
    v16bf a;
    const int kb = kk * 32 + 16 * half;
    #pragma unroll
    for (int i = 0; i < 8; ++i) {
      a[i]     = (bf16)arow[kb + i];
      a[i + 8] = (bf16)arow[kb + 8 + i];
    }
    return a;
  };

  v8f acc[4];
  #pragma unroll
  for (int nt = 0; nt < 4; ++nt)
    #pragma unroll
    for (int j = 0; j < 8; ++j) acc[nt][j] = 0.0f;

  const int NK = Dc / 32;
  v16bf a_cur = loadA(0);
  v16bf b_cur[4];
  #pragma unroll
  for (int nt = 0; nt < 4; ++nt)
    b_cur[nt] = *(const v16bf*)(wbase + (size_t)nt * 16 * Dc);

  for (int kk = 0; kk < NK - 1; ++kk) {
    v16bf a_nxt = loadA(kk + 1);
    v16bf b_nxt[4];
    #pragma unroll
    for (int nt = 0; nt < 4; ++nt)
      b_nxt[nt] = *(const v16bf*)(wbase + (size_t)nt * 16 * Dc + (kk + 1) * 32);
    #pragma unroll
    for (int nt = 0; nt < 4; ++nt)
      acc[nt] = wmma_bf16(a_cur, b_cur[nt], acc[nt]);
    a_cur = a_nxt;
    #pragma unroll
    for (int nt = 0; nt < 4; ++nt) b_cur[nt] = b_nxt[nt];
  }
  #pragma unroll
  for (int nt = 0; nt < 4; ++nt)
    acc[nt] = wmma_bf16(a_cur, b_cur[nt], acc[nt]);

  // Straight-line epilogue: single per-lane base, immediate offsets.
  if (TR) {
    // out[b][h][col][row], col = nt*16+n16 (stride S), row = mt*16+r+8*half
    bf16* obase = out + (((size_t)b * Hc + h) * Dh + n16) * Sc
                      + mt * 16 + 8 * half;
    #pragma unroll
    for (int nt = 0; nt < 4; ++nt)
      #pragma unroll
      for (int r = 0; r < 8; ++r)
        obase[(size_t)nt * 16 * Sc + r] = (bf16)(acc[nt][r] * scale);
  } else {
    // out[b][h][row][col]
    bf16* obase = out + (((size_t)b * Hc + h) * Sc + mt * 16 + 8 * half) * Dh
                      + n16;
    #pragma unroll
    for (int nt = 0; nt < 4; ++nt)
      #pragma unroll
      for (int r = 0; r < 8; ++r)
        obase[(size_t)r * Dh + nt * 16] = (bf16)(acc[nt][r] * scale);
  }
}

// ---------------------------------------------------------------------------
// Kernel 2: flash attention. One wave per 16-query tile per (b,h).
// q[b][h][s][dh], k[b][h][s][dh], vT[b][h][dh][s]  (all bf16, q pre-scaled).
// Output x[b][s][h*Dh+dh] bf16.
// ---------------------------------------------------------------------------
__global__ __launch_bounds__(32)
void mha_attn(const bf16* __restrict__ qb, const bf16* __restrict__ kb,
              const bf16* __restrict__ vtb, const int* __restrict__ pad,
              bf16* __restrict__ xb) {
  __shared__ bf16 pbuf[16 * 32];

  const int wid  = blockIdx.x;
  const int MT   = Sc / 16;
  const int mt   = wid % MT;
  const int h    = (wid / MT) % Hc;
  const int b    = wid / (MT * Hc);
  const int lane = threadIdx.x;
  const int n16  = lane & 15;
  const int half = lane >> 4;

  const size_t headoff = ((size_t)b * Hc + h) * (size_t)Sc * Dh;
  const bf16* qrow  = qb  + headoff + (size_t)(mt * 16 + n16) * Dh;
  // per-lane bases: k rows for the two key subtiles; v cols for the 4 nt tiles
  const bf16* kln   = kb  + headoff + (size_t)n16 * Dh + 16 * half;
  const bf16* vln   = vtb + headoff + (size_t)n16 * Sc + 16 * half;

  // q A-fragments for the two Dh chunks (K=0..31, 32..63)
  v16bf aq[2];
  aq[0] = load_a_bf16(qrow, half);
  aq[1] = load_a_bf16(qrow + 32, half);

  float m_[8], l_[8];
  v8f o[4];
  #pragma unroll
  for (int r = 0; r < 8; ++r) { m_[r] = -1.0e30f; l_[r] = 0.0f; }
  #pragma unroll
  for (int nt = 0; nt < 4; ++nt)
    #pragma unroll
    for (int j = 0; j < 8; ++j) o[nt][j] = 0.0f;

  const int vlen = Sc - pad[b];
  const int q0   = mt * 16;
  const int nkt  = (q0 + 15) / 32 + 1;          // causal truncation (<= S/32)

  for (int kt = 0; kt < nkt; ++kt) {
    const int key0 = kt * 32;

    // ---- batch-load all 4 K B-fragments, then 4 score WMMAs --------------
    // fragment (j = key subtile, c2 = dh chunk): row = key0 + j*16 + n16
    v16bf kf[2][2];
    #pragma unroll
    for (int j = 0; j < 2; ++j)
      #pragma unroll
      for (int c2 = 0; c2 < 2; ++c2)
        kf[j][c2] = *(const v16bf*)(kln + (size_t)(key0 + j * 16) * Dh
                                        + c2 * 32);
    v8f c0, c1;
    #pragma unroll
    for (int t = 0; t < 8; ++t) { c0[t] = 0.0f; c1[t] = 0.0f; }
    c0 = wmma_bf16(aq[0], kf[0][0], c0);
    c0 = wmma_bf16(aq[1], kf[0][1], c0);
    c1 = wmma_bf16(aq[0], kf[1][0], c1);
    c1 = wmma_bf16(aq[1], kf[1][1], c1);

    // ---- issue V B-fragment loads early (independent of softmax) ---------
    v16bf vf[4];
    #pragma unroll
    for (int nt = 0; nt < 4; ++nt)
      vf[nt] = *(const v16bf*)(vln + (size_t)nt * 16 * Sc + key0);

    // ---- mask + online softmax stats (hides v-load latency) --------------
    float s0[8], s1[8];
    #pragma unroll
    for (int r = 0; r < 8; ++r) { s0[r] = c0[r]; s1[r] = c1[r]; }
    const int col0 = key0 + n16;
    const int col1 = key0 + 16 + n16;
    #pragma unroll
    for (int r = 0; r < 8; ++r) {
      const int row = q0 + r + 8 * half;
      if (col0 > row || col0 >= vlen) s0[r] += NEGV;
      if (col1 > row || col1 >= vlen) s1[r] += NEGV;

      float tmax = redmax16(fmaxf(s0[r], s1[r]));
      float nm   = fmaxf(m_[r], tmax);
      float al   = __expf(m_[r] - nm);
      m_[r] = nm;
      float p0 = __expf(s0[r] - nm);
      float p1 = __expf(s1[r] - nm);
      l_[r] = l_[r] * al + redsum16(p0 + p1);
      #pragma unroll
      for (int nt = 0; nt < 4; ++nt) o[nt][r] *= al;

      // stage probabilities (C-layout -> row-major 16x32 tile in LDS)
      pbuf[(r + 8 * half) * 32 + n16]      = (bf16)p0;
      pbuf[(r + 8 * half) * 32 + 16 + n16] = (bf16)p1;
    }
    __syncthreads();

    // ---- reload P in A-layout, accumulate P @ V --------------------------
    v16bf pa = load_a_bf16(&pbuf[n16 * 32], half);
    __syncthreads();

    #pragma unroll
    for (int nt = 0; nt < 4; ++nt)
      o[nt] = wmma_bf16(pa, vf[nt], o[nt]);
  }

  // ---- normalize and store x[b][s][h*Dh + dh] ----------------------------
  float inv[8];
  #pragma unroll
  for (int r = 0; r < 8; ++r) inv[r] = 1.0f / l_[r];
  bf16* obase = xb + ((size_t)b * Sc + q0 + 8 * half) * Dc + h * Dh + n16;
  #pragma unroll
  for (int nt = 0; nt < 4; ++nt)
    #pragma unroll
    for (int r = 0; r < 8; ++r)
      obase[(size_t)r * Dc + nt * 16] = (bf16)(o[nt][r] * inv[r]);
}

// ---------------------------------------------------------------------------
// Kernel 3: output projection. out = x @ W_out + b_out  (f32 out).
// x bf16 [B*S][D], WoT bf16 [N][D]. One wave: 16 rows x 64 cols, pipelined.
// ---------------------------------------------------------------------------
__global__ __launch_bounds__(32)
void mha_outproj(const bf16* __restrict__ xb, const bf16* __restrict__ WoT,
                 const float* __restrict__ bias, float* __restrict__ out) {
  const int wid  = blockIdx.x;                 // (B*S/16) * (D/64)
  const int NT   = Dc / 64;                    // 16
  const int nt4  = wid % NT;
  const int mt   = wid / NT;
  const int lane = threadIdx.x;
  const int n16  = lane & 15;
  const int half = lane >> 4;
  const int m0   = mt * 16;
  const int n0   = nt4 * 64;

  const bf16* arow  = xb  + (size_t)(m0 + n16) * Dc;
  const bf16* wbase = WoT + (size_t)(n0 + n16) * Dc + 16 * half;

  v8f acc[4];
  #pragma unroll
  for (int nt = 0; nt < 4; ++nt)
    #pragma unroll
    for (int j = 0; j < 8; ++j) acc[nt][j] = 0.0f;

  const int NK = Dc / 32;
  v16bf a_cur = load_a_bf16(arow, half);
  v16bf b_cur[4];
  #pragma unroll
  for (int nt = 0; nt < 4; ++nt)
    b_cur[nt] = *(const v16bf*)(wbase + (size_t)nt * 16 * Dc);

  for (int kk = 0; kk < NK - 1; ++kk) {
    v16bf a_nxt = load_a_bf16(arow + (kk + 1) * 32, half);
    v16bf b_nxt[4];
    #pragma unroll
    for (int nt = 0; nt < 4; ++nt)
      b_nxt[nt] = *(const v16bf*)(wbase + (size_t)nt * 16 * Dc + (kk + 1) * 32);
    #pragma unroll
    for (int nt = 0; nt < 4; ++nt)
      acc[nt] = wmma_bf16(a_cur, b_cur[nt], acc[nt]);
    a_cur = a_nxt;
    #pragma unroll
    for (int nt = 0; nt < 4; ++nt) b_cur[nt] = b_nxt[nt];
  }
  #pragma unroll
  for (int nt = 0; nt < 4; ++nt)
    acc[nt] = wmma_bf16(a_cur, b_cur[nt], acc[nt]);

  float* obase = out + (size_t)(m0 + 8 * half) * Dc + n0 + n16;
  #pragma unroll
  for (int nt = 0; nt < 4; ++nt) {
    const float bb = bias[n0 + nt * 16 + n16];
    #pragma unroll
    for (int r = 0; r < 8; ++r)
      obase[(size_t)r * Dc + nt * 16] = acc[nt][r] + bb;
  }
}

// ---------------------------------------------------------------------------
extern "C" void kernel_launch(void* const* d_in, const int* in_sizes, int n_in,
                              void* d_out, int out_size, void* d_ws, size_t ws_size,
                              hipStream_t stream) {
  const float* Q   = (const float*)d_in[0];
  const float* K   = (const float*)d_in[1];
  const float* V   = (const float*)d_in[2];
  const int*   pad = (const int*)  d_in[3];
  const float* Wq  = (const float*)d_in[4];
  const float* Wk  = (const float*)d_in[5];
  const float* Wv  = (const float*)d_in[6];
  const float* Wo  = (const float*)d_in[7];
  const float* bo  = (const float*)d_in[8];
  float* out = (float*)d_out;

  char* ws = (char*)d_ws;
  size_t off = 0;
  auto carve = [&](size_t bytes) {
    void* p = ws + off;
    off += (bytes + 255) & ~(size_t)255;
    return p;
  };
  const size_t wsz   = (size_t)Hc * Dh * Dc * sizeof(bf16);      // 2 MB each
  const size_t qkvsz = (size_t)Bc * Hc * Sc * Dh * sizeof(bf16); // 8 MB each
  const size_t xsz   = (size_t)Bc * Sc * Dc * sizeof(bf16);      // 8 MB

  bf16* WqT  = (bf16*)carve(wsz);
  bf16* WkT  = (bf16*)carve(wsz);
  bf16* WvT  = (bf16*)carve(wsz);
  bf16* WoT  = (bf16*)carve((size_t)Dc * Dc * sizeof(bf16));
  bf16* qbuf = (bf16*)carve(qkvsz);
  bf16* kbuf = (bf16*)carve(qkvsz);
  bf16* vbuf = (bf16*)carve(qkvsz);  // transposed [b][h][Dh][S]
  bf16* xbuf = (bf16*)carve(xsz);

  // 0) weight prep (transpose + bf16 convert)
  {
    int n = Hc * Dc * Dh;
    int g = (n + 255) / 256;
    mha_wprep<<<g, 256, 0, stream>>>(Wq, WqT, Dc, Dh, Hc);
    mha_wprep<<<g, 256, 0, stream>>>(Wk, WkT, Dc, Dh, Hc);
    mha_wprep<<<g, 256, 0, stream>>>(Wv, WvT, Dc, Dh, Hc);
    int n2 = Dc * Dc;
    mha_wprep<<<(n2 + 255) / 256, 256, 0, stream>>>(Wo, WoT, Dc, Dc, 1);
  }

  // 1) projections (q pre-scaled by 1/sqrt(S); V stored transposed)
  const int projGrid = Bc * Hc * (Sc / 16);     // 4096 waves
  mha_proj<0><<<projGrid, 32, 0, stream>>>(Q, WqT, qbuf, QSCALE);
  mha_proj<0><<<projGrid, 32, 0, stream>>>(K, WkT, kbuf, 1.0f);
  mha_proj<1><<<projGrid, 32, 0, stream>>>(V, WvT, vbuf, 1.0f);

  // 2) flash attention
  mha_attn<<<projGrid, 32, 0, stream>>>(qbuf, kbuf, vbuf, pad, xbuf);

  // 3) output projection + bias
  const int opGrid = (Bc * Sc / 16) * (Dc / 64); // 4096 waves
  mha_outproj<<<opGrid, 32, 0, stream>>>(xbuf, WoT, bo, out);
}
